// GNN_node_Virtualnode_83391085019186
// MI455X (gfx1250) — matrix operand order repaired and checked by reading, not compiled, not run on hardware
//
#include <hip/hip_runtime.h>
#include <hip/hip_bf16.h>

// ---------------- static problem config (matches reference) ----------------
#define N_NODES 100000
#define N_EDGES 800000
#define N_GRAPH 4096
#define DIM     256
#define NLAYERS 5

typedef __attribute__((ext_vector_type(16))) __bf16 v16bf;
typedef __attribute__((ext_vector_type(8)))  float  v8f;

// exclusive cumsum of OGB atom feature dims {119,4,12,12,10,6,6,2,2}
__constant__ int c_offs[9] = {0, 119, 123, 135, 147, 157, 163, 169, 171};

// ---------------------------------------------------------------------------
// AtomEncoder: h[n,c] = sum_j table[x[n,3+j]+off[j%9], c]   (18 lookups)
// ---------------------------------------------------------------------------
__global__ __launch_bounds__(256) void atom_encoder(const int* __restrict__ x,
                                                    const float* __restrict__ table,
                                                    float* __restrict__ h) {
    const int n = blockIdx.x;
    const int c = threadIdx.x;
    __shared__ int sidx[18];
    if (c < 18) sidx[c] = x[n * 21 + 3 + c] + c_offs[c % 9];
    __syncthreads();
    float s = 0.f;
#pragma unroll
    for (int j = 0; j < 18; ++j) s += table[(size_t)sidx[j] * DIM + c];
    h[(size_t)n * DIM + c] = s;
}

// ---------------------------------------------------------------------------
// h_in = h + vn[batch[row]]
// ---------------------------------------------------------------------------
__global__ __launch_bounds__(256) void add_vn(const float* __restrict__ h,
                                              const float* __restrict__ vn,
                                              const int* __restrict__ batch,
                                              float* __restrict__ hin) {
    size_t i = (size_t)blockIdx.x * 256 + threadIdx.x;
    if (i >= (size_t)N_NODES * DIM) return;
    const int row = (int)(i >> 8);
    const int c   = (int)(i & 255);
    hin[i] = h[i] + vn[(size_t)batch[row] * DIM + c];
}

// ---------------------------------------------------------------------------
// generic fill-zero / copy (grid-stride)
// ---------------------------------------------------------------------------
__global__ void fillz(float* __restrict__ p, size_t n) {
    size_t i = (size_t)blockIdx.x * blockDim.x + threadIdx.x;
    size_t st = (size_t)gridDim.x * blockDim.x;
    for (; i < n; i += st) p[i] = 0.f;
}
__global__ void copyk(const float* __restrict__ a, float* __restrict__ b, size_t n) {
    size_t i = (size_t)blockIdx.x * blockDim.x + threadIdx.x;
    size_t st = (size_t)gridDim.x * blockDim.x;
    for (; i < n; i += st) b[i] = a[i];
}

// ---------------------------------------------------------------------------
// one-time weight transpose + f32 -> bf16 convert:  Wt[mat][n][k] = W[mat][k][n]
// ---------------------------------------------------------------------------
__global__ __launch_bounds__(256) void wcvt(const float* __restrict__ W,
                                            __bf16* __restrict__ Wt, int nmat) {
    size_t i = (size_t)blockIdx.x * 256 + threadIdx.x;
    if (i >= (size_t)nmat * DIM * DIM) return;
    const int mat = (int)(i >> 16);
    const int k   = (int)((i >> 8) & 255);
    const int n   = (int)(i & 255);
    Wt[((size_t)mat << 16) + ((size_t)n << 8) + k] = (__bf16)W[i];
}

// ---------------------------------------------------------------------------
// edge scatter: agg[dst] += relu(h_in[src]); one thread per (edge, 4-col chunk)
// ---------------------------------------------------------------------------
__global__ __launch_bounds__(256) void edge_scatter(const float* __restrict__ hin,
                                                    const int* __restrict__ ei,
                                                    float* __restrict__ agg) {
    long long idx = (long long)blockIdx.x * 256 + threadIdx.x;
    if (idx >= (long long)N_EDGES * 64) return;
    const int e  = (int)(idx >> 6);
    const int c4 = (int)(idx & 63) * 4;
    const int s = ei[e];
    const int d = ei[N_EDGES + e];
    const float4 v = *(const float4*)(hin + (size_t)s * DIM + c4);
    float* dst = agg + (size_t)d * DIM + c4;
    atomicAdd(dst + 0, fmaxf(v.x, 0.f));
    atomicAdd(dst + 1, fmaxf(v.y, 0.f));
    atomicAdd(dst + 2, fmaxf(v.z, 0.f));
    atomicAdd(dst + 3, fmaxf(v.w, 0.f));
}

// ---------------------------------------------------------------------------
// graph pool scatter: pooled[batch[row]] += h_in[row]   (pooled preloaded = vn)
// ---------------------------------------------------------------------------
__global__ __launch_bounds__(256) void pool_scatter(const float* __restrict__ hin,
                                                    const int* __restrict__ batch,
                                                    float* __restrict__ pooled) {
    size_t i = (size_t)blockIdx.x * 256 + threadIdx.x;
    if (i >= (size_t)N_NODES * DIM) return;
    const int row = (int)(i >> 8);
    const int c   = (int)(i & 255);
    atomicAdd(&pooled[(size_t)batch[row] * DIM + c], hin[i]);
}

// ---------------------------------------------------------------------------
// vn init broadcast
// ---------------------------------------------------------------------------
__global__ void bcast_vn(const float* __restrict__ vn0, float* __restrict__ vn) {
    size_t i = (size_t)blockIdx.x * 256 + threadIdx.x;
    if (i >= (size_t)N_GRAPH * DIM) return;
    vn[i] = vn0[i & 255];
}

// ---------------------------------------------------------------------------
// BatchNorm stats: st[0..255]=sum, st[256..511]=sumsq (column-wise, atomics)
// ---------------------------------------------------------------------------
__global__ __launch_bounds__(256) void bn_stats(const float* __restrict__ X,
                                                float* __restrict__ st,
                                                int M, int rowsPerBlock) {
    const int col = threadIdx.x;
    int r0 = blockIdx.x * rowsPerBlock;
    int r1 = r0 + rowsPerBlock; if (r1 > M) r1 = M;
    float s = 0.f, q = 0.f;
    for (int r = r0; r < r1; ++r) {
        float v = X[(size_t)r * DIM + col];
        s += v; q += v * v;
    }
    atomicAdd(&st[col], s);
    atomicAdd(&st[DIM + col], q);
}

// ---------------------------------------------------------------------------
// BatchNorm apply (training-mode, biased var), optional relu
// ---------------------------------------------------------------------------
__global__ __launch_bounds__(256) void bn_apply(const float* __restrict__ X,
                                                const float* __restrict__ st,
                                                const float* __restrict__ g,
                                                const float* __restrict__ b,
                                                float* __restrict__ O,
                                                int M, int relu) {
    size_t i = (size_t)blockIdx.x * 256 + threadIdx.x;
    if (i >= (size_t)M * DIM) return;
    const int c = (int)(i & 255);
    const float inv = 1.0f / (float)M;
    const float m  = st[c] * inv;
    const float vv = st[DIM + c] * inv - m * m;
    float y = (X[i] - m) * rsqrtf(vv + 1e-5f) * g[c] + b[c];
    if (relu) y = fmaxf(y, 0.f);
    O[i] = y;
}

// ---------------------------------------------------------------------------
// WMMA GEMM:  C[M,256] = actA(X[,Y]) @ W[256,256] + bias
//   MODE 0: A = X    MODE 1: A = relu(X)    MODE 2: A = (1+eps)*X + Y
//
// Block = 256 threads (8 waves), 64x64 output tile; wave w computes tiles
// (m = w%4, n-cols = (w>>2)*32 + {0,16}); K swept in 32-wide bf16 slabs via
// v_wmma_f32_16x16x32_bf16 (f32 accumulation).
//
// Fast fragment I/O:
//  * A tile staged in LDS in FRAGMENT-MAJOR layout: element (m,k) lives at
//    lane = 16*(k>>4) + (m&15), idx = k&15.  A staging thread
//    loading 8 consecutive k writes one contiguous uint4 (8 bf16); each lane
//    reads its whole v16bf fragment as two ds_load_b128.
//  * B comes from the pre-transposed bf16 weights Wt[n][k]: lane (n=lane&15,
//    half=lane>>4) reads 16 contiguous bf16 = two global_load_b128 (L2-hot,
//    2.4 MB of weights total vs 192 MB L2) -- no LDS, no element shuffles.
// ---------------------------------------------------------------------------
template <int MODE>
__global__ __launch_bounds__(256) void gemm256(const float* __restrict__ X,
                                               const float* __restrict__ Y,
                                               const float* __restrict__ eps_ptr,
                                               const __bf16* __restrict__ Wt,
                                               const float* __restrict__ bias,
                                               float* __restrict__ C, int M) {
    // AsF[mtile][lane][part]: part0 = frag elems 0..7, part1 = elems 8..15
    __shared__ uint4 AsF[4][32][2];          // 4 KB

    const int t     = threadIdx.x;
    const int wave  = t >> 5;
    const int lane  = t & 31;
    const int half  = lane >> 4;
    const int ln    = lane & 15;
    const int mTile = wave & 3;              // 0..3 -> 16-row tile
    const int nPair = wave >> 2;             // 0..1 -> 32-col pair
    const int mBase = blockIdx.x * 64;
    const int nBase = blockIdx.y * 64;

    float alpha = 1.0f;
    if (MODE == 2) alpha = 1.0f + eps_ptr[0];

    // staging coords (per thread): 8 consecutive k of one row
    const int srow = t >> 2;                 // 0..63
    const int sc0  = (t & 3) * 8;            // 0/8/16/24
    const int dmt   = srow >> 4;
    const int dlane = (sc0 >> 4) * 16 + (srow & 15);
    const int dpart = (sc0 >> 3) & 1;
    const int sgrow = mBase + srow;

    // per-lane B pointers (bf16, n-major): row n, 16 contiguous k per frag
    const int nA = nBase + nPair * 32 + ln;
    const int nB = nA + 16;
    const __bf16* wpA = Wt + ((size_t)nA << 8) + half * 16;
    const __bf16* wpB = Wt + ((size_t)nB << 8) + half * 16;

    v8f acc0 = {}; v8f acc1 = {};

    for (int kk = 0; kk < DIM; kk += 32) {
        // ---- stage A tile (fused epilogue, packed to bf16 uint4) ----
        {
            float tmp[8];
            if (sgrow < M) {
                const float* xp = X + (size_t)sgrow * DIM + kk + sc0;
#pragma unroll
                for (int i = 0; i < 8; ++i) tmp[i] = xp[i];
                if (MODE == 1) {
#pragma unroll
                    for (int i = 0; i < 8; ++i) tmp[i] = fmaxf(tmp[i], 0.f);
                }
                if (MODE == 2) {
                    const float* yp = Y + (size_t)sgrow * DIM + kk + sc0;
#pragma unroll
                    for (int i = 0; i < 8; ++i) tmp[i] = alpha * tmp[i] + yp[i];
                }
            } else {
#pragma unroll
                for (int i = 0; i < 8; ++i) tmp[i] = 0.f;
            }
            union { __bf16 h[8]; uint4 u; } pk;
#pragma unroll
            for (int i = 0; i < 8; ++i) pk.h[i] = (__bf16)tmp[i];
            AsF[dmt][dlane][dpart] = pk.u;
        }
        __syncthreads();

        // ---- A fragment: two ds_load_b128 ----
        union { uint4 q[2]; v16bf v; } fa;
        fa.q[0] = AsF[mTile][lane][0];
        fa.q[1] = AsF[mTile][lane][1];

        // ---- B fragments: two global_load_b128 each (L2-resident weights) ----
        union { uint4 q[2]; v16bf v; } fb0, fb1;
        {
            const uint4* p0 = (const uint4*)(wpA + kk);
            const uint4* p1 = (const uint4*)(wpB + kk);
            fb0.q[0] = p0[0]; fb0.q[1] = p0[1];
            fb1.q[0] = p1[0]; fb1.q[1] = p1[1];
        }

        acc0 = __builtin_amdgcn_wmma_f32_16x16x32_bf16(false, fa.v, false, fb0.v,
                                                       (short)0, acc0, false, false);
        acc1 = __builtin_amdgcn_wmma_f32_16x16x32_bf16(false, fa.v, false, fb1.v,
                                                       (short)0, acc1, false, false);
        __syncthreads();
    }

    // ---- write back (+bias); C/D layout: row = half*8 + r, col = lane&15 ----
#pragma unroll
    for (int r = 0; r < 8; ++r) {
        const int row = mBase + mTile * 16 + half * 8 + r;
        if (row < M) {
            C[(size_t)row * DIM + nA] = acc0[r] + bias[nA];
            C[(size_t)row * DIM + nB] = acc1[r] + bias[nB];
        }
    }
}

// ---------------------------------------------------------------------------
extern "C" void kernel_launch(void* const* d_in, const int* in_sizes, int n_in,
                              void* d_out, int out_size, void* d_ws, size_t ws_size,
                              hipStream_t stream) {
    const int*   x     = (const int*)d_in[0];
    const int*   ei    = (const int*)d_in[1];
    const int*   batch = (const int*)d_in[2];
    const float* table = (const float*)d_in[3];
    const float* vn0   = (const float*)d_in[4];
    const float* eps   = (const float*)d_in[5];
    const float* gW1   = (const float*)d_in[6];
    const float* gb1   = (const float*)d_in[7];
    const float* gg1   = (const float*)d_in[8];
    const float* gbe1  = (const float*)d_in[9];
    const float* gW2   = (const float*)d_in[10];
    const float* gb2   = (const float*)d_in[11];
    const float* bng   = (const float*)d_in[12];
    const float* bnb   = (const float*)d_in[13];
    const float* vW1   = (const float*)d_in[14];
    const float* vb1   = (const float*)d_in[15];
    const float* vg1   = (const float*)d_in[16];
    const float* vbe1  = (const float*)d_in[17];
    const float* vW2   = (const float*)d_in[18];
    const float* vb2   = (const float*)d_in[19];
    const float* vg2   = (const float*)d_in[20];
    const float* vbe2  = (const float*)d_in[21];
    float* out = (float*)d_out;

    const size_t NB  = (size_t)N_NODES * DIM;   // 25.6M floats
    const size_t GB  = (size_t)N_GRAPH * DIM;   // 1.05M floats
    const size_t WKW = (size_t)DIM * DIM;       // 65536
    float* ws     = (float*)d_ws;
    float* h      = ws;                 // node features
    float* hin    = ws + NB;            // h + vn[batch]
    float* agg    = ws + 2 * NB;        // edge aggregation / bn scratch
    float* t1     = ws + 3 * NB;        // gemm output scratch
    float* vnA    = ws + 4 * NB;
    float* vnB    = vnA + GB;
    float* pooled = vnB + GB;
    float* vt1    = pooled + GB;
    float* vt2    = vt1 + GB;
    float* st     = vt2 + GB;           // 512 floats of BN stats
    __bf16* wt    = (__bf16*)(st + 512);   // 18 * 65536 bf16 (transposed)
    __bf16* gW1t  = wt;
    __bf16* gW2t  = wt + 5 * WKW;
    __bf16* vW1t  = wt + 10 * WKW;
    __bf16* vW2t  = wt + 14 * WKW;

    dim3 gemmGridN((N_NODES + 63) / 64, 4);
    dim3 gemmGridG((N_GRAPH + 63) / 64, 4);
    const int nbElem = (int)(NB / 256);                       // 100000 blocks
    const int gbElem = (int)(GB / 256);                       // 4096 blocks
    const int edgeBlocks = (int)(((long long)N_EDGES * 64 + 255) / 256);
    const int statN = (N_NODES + 511) / 512;
    const int statG = (N_GRAPH + 511) / 512;
    const int wblk5 = (int)((5 * WKW + 255) / 256);
    const int wblk4 = (int)((4 * WKW + 255) / 256);

    // one-time weight transpose/convert (bf16, n-major; 2.4 MB -> L2-hot)
    wcvt<<<wblk5, 256, 0, stream>>>(gW1, gW1t, 5);
    wcvt<<<wblk5, 256, 0, stream>>>(gW2, gW2t, 5);
    wcvt<<<wblk4, 256, 0, stream>>>(vW1, vW1t, 4);
    wcvt<<<wblk4, 256, 0, stream>>>(vW2, vW2t, 4);

    // encoder + virtual-node init
    atom_encoder<<<N_NODES, 256, 0, stream>>>(x, table, h);
    bcast_vn<<<gbElem, 256, 0, stream>>>(vn0, vnA);

    float* vncur = vnA;
    float* vnnxt = vnB;

    for (int l = 0; l < NLAYERS; ++l) {
        // h_in = h + vn[batch]
        add_vn<<<nbElem, 256, 0, stream>>>(h, vncur, batch, hin);

        // agg = segment_sum(relu(h_in[src]) -> dst)
        fillz<<<2048, 256, 0, stream>>>(agg, NB);
        edge_scatter<<<edgeBlocks, 256, 0, stream>>>(hin, ei, agg);

        // t1 = ((1+eps)*h_in + agg) @ W1 + b1
        gemm256<2><<<gemmGridN, 256, 0, stream>>>(hin, agg, eps + l,
                                                  gW1t + l * WKW,
                                                  gb1 + l * DIM, t1, N_NODES);
        // z = BN(t1)  (into agg)
        fillz<<<2, 256, 0, stream>>>(st, 512);
        bn_stats<<<statN, 256, 0, stream>>>(t1, st, N_NODES, 512);
        bn_apply<<<nbElem, 256, 0, stream>>>(t1, st, gg1 + l * DIM, gbe1 + l * DIM,
                                             agg, N_NODES, 0);
        // t1 = relu(z) @ W2 + b2
        gemm256<1><<<gemmGridN, 256, 0, stream>>>(agg, nullptr, nullptr,
                                                  gW2t + l * WKW,
                                                  gb2 + l * DIM, t1, N_NODES);
        // h = BN(t1) [+relu unless last layer]
        fillz<<<2, 256, 0, stream>>>(st, 512);
        bn_stats<<<statN, 256, 0, stream>>>(t1, st, N_NODES, 512);
        const int last = (l == NLAYERS - 1);
        bn_apply<<<nbElem, 256, 0, stream>>>(t1, st, bng + l * DIM, bnb + l * DIM,
                                             last ? out : h, N_NODES, last ? 0 : 1);

        if (!last) {
            // pooled = vn + segment_sum(h_in by batch)
            copyk<<<256, 256, 0, stream>>>(vncur, pooled, GB);
            pool_scatter<<<nbElem, 256, 0, stream>>>(hin, batch, pooled);
            // t = relu(BN(pooled @ vW1 + vb1))
            gemm256<0><<<gemmGridG, 256, 0, stream>>>(pooled, nullptr, nullptr,
                                                      vW1t + l * WKW,
                                                      vb1 + l * DIM, vt1, N_GRAPH);
            fillz<<<2, 256, 0, stream>>>(st, 512);
            bn_stats<<<statG, 256, 0, stream>>>(vt1, st, N_GRAPH, 512);
            bn_apply<<<gbElem, 256, 0, stream>>>(vt1, st, vg1 + l * DIM, vbe1 + l * DIM,
                                                 vt2, N_GRAPH, 1);
            // vn = relu(BN(t @ vW2 + vb2))
            gemm256<0><<<gemmGridG, 256, 0, stream>>>(vt2, nullptr, nullptr,
                                                      vW2t + l * WKW,
                                                      vb2 + l * DIM, vt1, N_GRAPH);
            fillz<<<2, 256, 0, stream>>>(st, 512);
            bn_stats<<<statG, 256, 0, stream>>>(vt1, st, N_GRAPH, 512);
            bn_apply<<<gbElem, 256, 0, stream>>>(vt1, st, vg2 + l * DIM, vbe2 + l * DIM,
                                                 vnnxt, N_GRAPH, 1);
            float* tmp = vncur; vncur = vnnxt; vnnxt = tmp;
        }
    }
    (void)in_sizes; (void)n_in; (void)out_size; (void)ws_size;
}
